// LSTMActorCritic_88665304858858
// MI455X (gfx1250) — compile-verified, matrix operand order checked by reading, not run on hardware
//
#include <hip/hip_runtime.h>
#include <hip/hip_bf16.h>

#define NSEQD 256
#define TTD   1024
#define OBSD  128
#define ACTD  16
#define HD    64
#define NTOK  (NSEQD * TTD)

#define XHS 136   // scan [feat|h] LDS row stride (halves)  -> conflict-free frags
#define GS  260   // scan gates LDS row stride (floats)
#define STS 136   // mlp staged obs tile row stride (halves)
#define TBS 72    // mlp transpose buffer row stride (halves)

typedef __attribute__((ext_vector_type(16))) _Float16 v16h;
typedef __attribute__((ext_vector_type(8)))  float    v8f;
typedef __attribute__((ext_vector_type(4)))  _Float16 v4h;
typedef __attribute__((ext_vector_type(4)))  float    v4f;

#define AS1 __attribute__((address_space(1)))
typedef const float    AS1* gfp;
typedef const _Float16 AS1* ghp;
typedef _Float16       AS1* ghpw;
typedef float          AS1* gfpw;
typedef const v4f      AS1* gf4p;
typedef const v4h      AS1* gv4cp;
typedef v4h            AS1* gv4p;
typedef const int      AS1* gip;

// ---------------------------------------------------------------- helpers ---

__device__ inline v8f v8f_zero() {
    v8f z;
#pragma unroll
    for (int i = 0; i < 8; ++i) z[i] = 0.0f;
    return z;
}

__device__ inline v8f wmma_f16(v16h a, v16h b, v8f c) {
    return __builtin_amdgcn_wmma_f32_16x16x32_f16(
        false, a, false, b, (short)0, c, false, false);
}

// A fragment (16x32 f16) from a row-major f16 LDS tile.
// Lane L: row = L&15; elems 0..7 <-> K = hi*8+j, elems 8..15 <-> K = 16+hi*8+j.
__device__ inline v16h a_frag_lds(const _Float16* base, int stride,
                                  int k0, int lane) {
    int row = lane & 15, hi = lane >> 4;
    const _Float16* p0 = base + row * stride + k0 + hi * 8;
    v16h a;
#pragma unroll
    for (int j = 0; j < 8; ++j) {
        a[j]     = p0[j];
        a[8 + j] = p0[16 + j];
    }
    return a;
}

// B fragment (32x16 f16) for Y = X @ W^T from pre-converted f16 weights
// in global memory: B[k][n] = W[n0+n][k0+k]; lane holds output column.
__device__ inline v16h b_frag_h(const _Float16* W, int Kd, int n0, int k0,
                                int lane) {
    int n = n0 + (lane & 15), hi = lane >> 4;
    ghp p0 = (ghp)(W + (size_t)n * Kd + k0 + hi * 8);
    v16h b;
#pragma unroll
    for (int j = 0; j < 8; ++j) {
        b[j]     = p0[j];
        b[8 + j] = p0[16 + j];
    }
    return b;
}

__device__ inline float sigf(float x) { return 1.0f / (1.0f + __expf(-x)); }

// ------------------------------------------------- weight f32->f16 pre-pass -

__global__ __launch_bounds__(256) void cvt_f32_to_f16(
    const float* __restrict__ src, _Float16* __restrict__ dst, int n) {
    int i = (blockIdx.x * 256 + threadIdx.x) * 4;
    if (i < n) {
        v4f v = *(gf4p)((gfp)src + i);
        v4h h;
#pragma unroll
        for (int q = 0; q < 4; ++q) h[q] = (_Float16)v[q];
        *(gv4p)((ghpw)dst + i) = h;
    }
}

// ------------------------------------------------------------ MLP kernel ----
// Per wave: 16 tokens. Coalesced f32 obs stage -> LDS f16, two WMMA layers
// with tanh, coalesced f16 feats writeback. Per-wave-private LDS tiles, so no
// block barriers are needed (DS ops are in-order within a wave).

__device__ inline void mlp_net(const v16h A1[4],
                               const _Float16* W1, const float* b1,
                               const _Float16* W2, const float* b2,
                               _Float16* fo /* feats + net offset */,
                               size_t t0, _Float16* tb, int lane) {
    const int col16 = lane & 15;
    const int m0    = (lane >> 4) * 8;

    // layer 1: 16x64, K=128
    v8f C[4];
#pragma unroll
    for (int nt = 0; nt < 4; ++nt) C[nt] = v8f_zero();
#pragma unroll
    for (int nt = 0; nt < 4; ++nt)
#pragma unroll
        for (int kc = 0; kc < 4; ++kc)
            C[nt] = wmma_f16(A1[kc], b_frag_h(W1, OBSD, nt * 16, kc * 32, lane),
                             C[nt]);
#pragma unroll
    for (int nt = 0; nt < 4; ++nt) {
        int   col = nt * 16 + col16;
        float bb  = b1[col];
#pragma unroll
        for (int r = 0; r < 8; ++r)
            tb[(m0 + r) * TBS + col] = (_Float16)tanhf(C[nt][r] + bb);
    }

    // layer 2: 16x64, K=64
    v16h A2[2];
#pragma unroll
    for (int kc = 0; kc < 2; ++kc) A2[kc] = a_frag_lds(tb, TBS, kc * 32, lane);

    v8f D[4];
#pragma unroll
    for (int nt = 0; nt < 4; ++nt) D[nt] = v8f_zero();
#pragma unroll
    for (int nt = 0; nt < 4; ++nt)
#pragma unroll
        for (int kc = 0; kc < 2; ++kc)
            D[nt] = wmma_f16(A2[kc], b_frag_h(W2, HD, nt * 16, kc * 32, lane),
                             D[nt]);

    // tanh -> tbuf (A2 already consumed), then coalesced writeback
#pragma unroll
    for (int nt = 0; nt < 4; ++nt) {
        int   col = nt * 16 + col16;
        float bb  = b2[col];
#pragma unroll
        for (int r = 0; r < 8; ++r)
            tb[(m0 + r) * TBS + col] = (_Float16)tanhf(D[nt][r] + bb);
    }
#pragma unroll
    for (int i = 0; i < 8; ++i) {
        int r  = i * 2 + (lane >> 4);
        v4h hv = *(const v4h*)(&tb[r * TBS + col16 * 4]);
        *(gv4p)((ghpw)(fo + (t0 + r) * HD) + col16 * 4) = hv;
    }
}

__global__ __launch_bounds__(256) void mlp_kernel(
    const float* __restrict__ obs,
    const _Float16* __restrict__ W1p, const float* __restrict__ bp1,
    const _Float16* __restrict__ W2p, const float* __restrict__ bp2,
    const _Float16* __restrict__ W1v, const float* __restrict__ bv1,
    const _Float16* __restrict__ W2v, const float* __restrict__ bv2,
    _Float16* __restrict__ feats)  // [2][NTOK][64] f16
{
    __shared__ _Float16 stile[8][16 * STS];
    __shared__ _Float16 tbuf [8][16 * TBS];

    const int tid  = threadIdx.x;
    const int lane = tid & 31;
    const int w    = tid >> 5;
    const size_t t0 = ((size_t)blockIdx.x * 8 + w) * 16;

    // stage the 16x128 obs tile: fully coalesced v4f loads -> f16 LDS
    gfp src = (gfp)(obs + t0 * OBSD);
#pragma unroll
    for (int r = 0; r < 16; ++r) {
        v4f v = *(gf4p)(src + r * OBSD + lane * 4);
        v4h h;
#pragma unroll
        for (int q = 0; q < 4; ++q) h[q] = (_Float16)v[q];
        *(v4h*)(&stile[w][r * STS + lane * 4]) = h;
    }

    v16h A1[4];
#pragma unroll
    for (int kc = 0; kc < 4; ++kc)
        A1[kc] = a_frag_lds(&stile[w][0], STS, kc * 32, lane);

    mlp_net(A1, W1p, bp1, W2p, bp2, feats,                      t0, &tbuf[w][0], lane);
    mlp_net(A1, W1v, bv1, W2v, bv2, feats + (size_t)NTOK * HD,  t0, &tbuf[w][0], lane);
}

// ----------------------------------------------------------- LSTM scan -----
// 32 independent blocks: blockIdx>>4 = pi/vf, blockIdx&15 = 16-seq slice.
// Per step: gates(16x256) = [feat|h](16x128) @ [Wih|Whh]^T (WMMA, B frags
// register-resident across all 1024 steps), then the fused cell update with
// c in registers. Two barriers per step on the critical path.

__global__ __launch_bounds__(256) void lstm_scan_kernel(
    const _Float16* __restrict__ feats,  // [2][NTOK][64]
    const float* __restrict__ ep,        // [NTOK]
    const float* __restrict__ h0_pi, const float* __restrict__ c0_pi,
    const float* __restrict__ h0_vf, const float* __restrict__ c0_vf,
    const _Float16* __restrict__ wih_pi, const _Float16* __restrict__ whh_pi,
    const float* __restrict__ bih_pi, const float* __restrict__ bhh_pi,
    const _Float16* __restrict__ wih_vf, const _Float16* __restrict__ whh_vf,
    const float* __restrict__ bih_vf, const float* __restrict__ bhh_vf,
    _Float16* __restrict__ lat)          // [2][NTOK][64]
{
    __shared__ _Float16 xh[16 * XHS];    // [row][feat(0..63) | h(64..127)]
    __shared__ float    gates[16 * GS];

    const int lstm = blockIdx.x >> 4;
    const int s0   = (blockIdx.x & 15) * 16;

    const _Float16* Wih = lstm ? wih_vf : wih_pi;
    const _Float16* Whh = lstm ? whh_vf : whh_pi;
    const float*    bih = lstm ? bih_vf : bih_pi;
    const float*    bhh = lstm ? bhh_vf : bhh_pi;
    const float*    h0  = lstm ? h0_vf : h0_pi;
    const float*    c0  = lstm ? c0_vf : c0_pi;
    ghp  f  = (ghp)(feats + (size_t)lstm * NTOK * HD);
    ghpw lo = (ghpw)(lat  + (size_t)lstm * NTOK * HD);

    const int tid  = threadIdx.x;
    const int lane = tid & 31;
    const int w    = tid >> 5;

    // Register-resident B fragments of [Wih|Whh]^T for 32 gate columns.
    v16h B[2][4];
#pragma unroll
    for (int nt = 0; nt < 2; ++nt) {
        int n0 = (w * 2 + nt) * 16;
#pragma unroll
        for (int kc = 0; kc < 4; ++kc) {
            const _Float16* Ws = (kc < 2) ? Wih : Whh;
            int kb = (kc < 2) ? kc * 32 : (kc - 2) * 32;
            B[nt][kc] = b_frag_h(Ws, HD, n0, kb, lane);
        }
    }

    // Element-wise mapping: row en (0..15), 4 cols from ej.
    const int en = tid >> 4;
    const int ej = (tid & 15) * 4;

    float bi[4], bf[4], bg[4], bo[4], c[4];
    {
        gfp pbih = (gfp)bih, pbhh = (gfp)bhh, pc0 = (gfp)c0, ph0 = (gfp)h0;
        v4h hinit;
#pragma unroll
        for (int q = 0; q < 4; ++q) {
            int j = ej + q;
            bi[q] = pbih[j]       + pbhh[j];
            bf[q] = pbih[64 + j]  + pbhh[64 + j];
            bg[q] = pbih[128 + j] + pbhh[128 + j];
            bo[q] = pbih[192 + j] + pbhh[192 + j];
            c[q]  = pc0[(s0 + en) * HD + j];
            hinit[q] = (_Float16)ph0[(s0 + en) * HD + j];
        }
        *(v4h*)(&xh[en * XHS + 64 + ej]) = hinit;
    }

    const int col16 = lane & 15;
    const int m0    = (lane >> 4) * 8;
    const int gc0   = (w * 2 + 0) * 16 + col16;
    const int gc1   = (w * 2 + 1) * 16 + col16;

    for (int t = 0; t < TTD; ++t) {
        // ---- phase 1: feat(t) -> LDS; episode reset mask on h and c ----
        const size_t tok = (size_t)(s0 + en) * TTD + t;
        const float  m   = 1.0f - ((gfp)ep)[tok];
        *(v4h*)(&xh[en * XHS + ej]) = *(gv4cp)(f + tok * HD + ej);
        {
            v4h hv = *(const v4h*)(&xh[en * XHS + 64 + ej]);
#pragma unroll
            for (int q = 0; q < 4; ++q) {
                c[q] *= m;
                hv[q] = (_Float16)((float)hv[q] * m);
            }
            *(v4h*)(&xh[en * XHS + 64 + ej]) = hv;
        }
        __builtin_prefetch(feats + (size_t)lstm * NTOK * HD + (tok + 1) * HD + ej, 0, 1);
        __syncthreads();  // barrier A

        // ---- phase 2: gates = [feat|h] @ [Wih|Whh]^T ----
        v8f acc0 = v8f_zero(), acc1 = v8f_zero();
#pragma unroll
        for (int kc = 0; kc < 4; ++kc) {
            v16h A = a_frag_lds(xh, XHS, kc * 32, lane);
            acc0 = wmma_f16(A, B[0][kc], acc0);
            acc1 = wmma_f16(A, B[1][kc], acc1);
        }
#pragma unroll
        for (int r = 0; r < 8; ++r) {
            gates[(m0 + r) * GS + gc0] = acc0[r];
            gates[(m0 + r) * GS + gc1] = acc1[r];
        }
        __syncthreads();  // barrier B

        // ---- phase 3: cell update; h -> LDS + global latents ----
        {
            v4f gi4 = *(const v4f*)(&gates[en * GS + ej]);
            v4f gf4 = *(const v4f*)(&gates[en * GS + 64 + ej]);
            v4f gg4 = *(const v4f*)(&gates[en * GS + 128 + ej]);
            v4f go4 = *(const v4f*)(&gates[en * GS + 192 + ej]);
            v4h hv;
#pragma unroll
            for (int q = 0; q < 4; ++q) {
                float cn = sigf(gf4[q] + bf[q]) * c[q] +
                           sigf(gi4[q] + bi[q]) * tanhf(gg4[q] + bg[q]);
                c[q] = cn;
                hv[q] = (_Float16)(sigf(go4[q] + bo[q]) * tanhf(cn));
            }
            *(v4h*)(&xh[en * XHS + 64 + ej]) = hv;
            *(gv4p)(lo + tok * HD + ej) = hv;
        }
        // no barrier here: phase-1(t+1) touches disjoint/same-thread cells,
        // and barrier A orders phase-3 gate reads vs next gate writes.
    }
}

// ------------------------------------------------------------- heads -------

__global__ __launch_bounds__(256) void head_kernel(
    const _Float16* __restrict__ lat,    // [2][NTOK][64]
    const int* __restrict__ action,
    const float* __restrict__ Wa, const float* __restrict__ ba,
    const float* __restrict__ Wc, const float* __restrict__ bc,
    float* __restrict__ out)             // [logprob N | entropy N | value N]
{
    __shared__ float    Was[ACTD * HD];
    __shared__ _Float16 hp[16 * HD];
    __shared__ _Float16 hv[16 * HD];

    const int tid = threadIdx.x;
    for (int i = tid; i < ACTD * HD; i += 256) Was[i] = ((gfp)Wa)[i];

    const int r = tid >> 4;
    const int a = tid & 15;
    const size_t idx = (size_t)blockIdx.x * 16 + r;

    ghp lp = (ghp)(lat + idx * HD);
    ghp lv = (ghp)(lat + (size_t)NTOK * HD + idx * HD);
    *(v4h*)(&hp[r * HD + a * 4]) = *(gv4cp)(lp + a * 4);
    *(v4h*)(&hv[r * HD + a * 4]) = *(gv4cp)(lv + a * 4);
    __syncthreads();

    float lg = ((gfp)ba)[a];
#pragma unroll
    for (int k = 0; k < HD; ++k)
        lg += (float)hp[r * HD + k] * Was[a * HD + k];

    float mx = lg;
#pragma unroll
    for (int off = 8; off > 0; off >>= 1)
        mx = fmaxf(mx, __shfl_xor(mx, off, 16));
    float s = __expf(lg - mx);
#pragma unroll
    for (int off = 8; off > 0; off >>= 1)
        s += __shfl_xor(s, off, 16);
    float logp = lg - mx - __logf(s);
    float p    = __expf(logp);

    float ent = -p * logp;
#pragma unroll
    for (int off = 8; off > 0; off >>= 1)
        ent += __shfl_xor(ent, off, 16);

    int   act = ((gip)action)[idx];
    float sel = (a == act) ? logp : 0.0f;
#pragma unroll
    for (int off = 8; off > 0; off >>= 1)
        sel += __shfl_xor(sel, off, 16);

    float pv = 0.0f;
#pragma unroll
    for (int q = 0; q < 4; ++q)
        pv += (float)hv[r * HD + a * 4 + q] * ((gfp)Wc)[a * 4 + q];
#pragma unroll
    for (int off = 8; off > 0; off >>= 1)
        pv += __shfl_xor(pv, off, 16);

    if (a == 0) {
        gfpw o = (gfpw)out;
        o[idx]            = sel;
        o[NTOK + idx]     = ent;
        o[2 * NTOK + idx] = pv + ((gfp)bc)[0];
    }
}

// ----------------------------------------------------------------- launch --

extern "C" void kernel_launch(void* const* d_in, const int* in_sizes, int n_in,
                              void* d_out, int out_size, void* d_ws, size_t ws_size,
                              hipStream_t stream) {
    const float* obs    = (const float*)d_in[0];
    const int*   action = (const int*)  d_in[1];
    const float* ep     = (const float*)d_in[2];
    const float* h_pi   = (const float*)d_in[3];
    const float* c_pi   = (const float*)d_in[4];
    const float* h_vf   = (const float*)d_in[5];
    const float* c_vf   = (const float*)d_in[6];
    const float* Wp1 = (const float*)d_in[7],  *bp1 = (const float*)d_in[8];
    const float* Wp2 = (const float*)d_in[9],  *bp2 = (const float*)d_in[10];
    const float* Wv1 = (const float*)d_in[11], *bv1 = (const float*)d_in[12];
    const float* Wv2 = (const float*)d_in[13], *bv2 = (const float*)d_in[14];
    const float* Wih_pi = (const float*)d_in[15], *Whh_pi = (const float*)d_in[16];
    const float* bih_pi = (const float*)d_in[17], *bhh_pi = (const float*)d_in[18];
    const float* Wih_vf = (const float*)d_in[19], *Whh_vf = (const float*)d_in[20];
    const float* bih_vf = (const float*)d_in[21], *bhh_vf = (const float*)d_in[22];
    const float* Wa = (const float*)d_in[23], *ba = (const float*)d_in[24];
    const float* Wc = (const float*)d_in[25], *bc = (const float*)d_in[26];

    // workspace layout (halves)
    _Float16* wsh  = (_Float16*)d_ws;
    _Float16* wp1h = wsh + 0;        // 64*128
    _Float16* wv1h = wsh + 8192;     // 64*128
    _Float16* wp2h = wsh + 16384;    // 64*64
    _Float16* wv2h = wsh + 20480;    // 64*64
    _Float16* wihp = wsh + 24576;    // 256*64
    _Float16* whhp = wsh + 40960;    // 256*64
    _Float16* wihv = wsh + 57344;    // 256*64
    _Float16* whhv = wsh + 73728;    // 256*64
    _Float16* feats = wsh + 90112;                      // 2*NTOK*64
    _Float16* lats  = feats + (size_t)2 * NTOK * HD;    // 2*NTOK*64
    float*    out   = (float*)d_out;

    cvt_f32_to_f16<<<8, 256, 0, stream>>>(Wp1, wp1h, 8192);
    cvt_f32_to_f16<<<8, 256, 0, stream>>>(Wv1, wv1h, 8192);
    cvt_f32_to_f16<<<4, 256, 0, stream>>>(Wp2, wp2h, 4096);
    cvt_f32_to_f16<<<4, 256, 0, stream>>>(Wv2, wv2h, 4096);
    cvt_f32_to_f16<<<16, 256, 0, stream>>>(Wih_pi, wihp, 16384);
    cvt_f32_to_f16<<<16, 256, 0, stream>>>(Whh_pi, whhp, 16384);
    cvt_f32_to_f16<<<16, 256, 0, stream>>>(Wih_vf, wihv, 16384);
    cvt_f32_to_f16<<<16, 256, 0, stream>>>(Whh_vf, whhv, 16384);

    mlp_kernel<<<NTOK / 128, 256, 0, stream>>>(
        obs, wp1h, bp1, wp2h, bp2, wv1h, bv1, wv2h, bv2, feats);

    lstm_scan_kernel<<<32, 256, 0, stream>>>(
        feats, ep, h_pi, c_pi, h_vf, c_vf,
        wihp, whhp, bih_pi, bhh_pi,
        wihv, whhv, bih_vf, bhh_vf, lats);

    head_kernel<<<NTOK / 16, 256, 0, stream>>>(
        lats, action, Wa, ba, Wc, bc, out);
}
